// Attention_55714315764415
// MI455X (gfx1250) — matrix operand order repaired and checked by reading, not compiled, not run on hardware
//
#include <hip/hip_runtime.h>

// ---------------------------------------------------------------------------
// Causal MHA forward for MI455X (gfx1250): bf16 WMMA with f32 accumulation.
// Pipeline: [qkv bf16 GEMM] -> [flash attention, wave-per-16-queries] -> [out GEMM]
// ---------------------------------------------------------------------------

typedef __attribute__((ext_vector_type(16))) __bf16 v16bf;
typedef __attribute__((ext_vector_type(8)))  __bf16 v8bf;
typedef __attribute__((ext_vector_type(8)))  float  v8f;

constexpr int Bsz  = 2;
constexpr int Tseq = 2048;
constexpr int Dm   = 2048;
constexpr int Hn   = 16;
constexpr int HSd  = 128;

#define CAT16(lo, hi) __builtin_shufflevector(lo, hi, 0,1,2,3,4,5,6,7,8,9,10,11,12,13,14,15)

static __device__ __forceinline__ v8f wmma_bf16(v16bf a, v16bf b, v8f c) {
  // v_wmma_f32_16x16x32_bf16  D = A(16x32) * B(32x16) + C
  return __builtin_amdgcn_wmma_f32_16x16x32_bf16(false, a, false, b, (short)0, c, false, false);
}

// A-fragment (16x32, 16-bit): lane holds row M=l%16; halves[0..7]=K[c0..c0+7],
// halves[8..15]=K[c0+16..c0+23], c0 = 8*(lane/16).  p points at (row, c0).
static __device__ __forceinline__ v16bf load_afrag(const __bf16* p) {
  v8bf lo = *(const v8bf*)p;
  v8bf hi = *(const v8bf*)(p + 16);
  return CAT16(lo, hi);
}
// B-fragment (32x16, 16-bit): lane holds col N=l%16; halves[0..7]=K[c0..c0+7],
// halves[8..15]=K[c0+16..c0+23], c0 = 8*(lane/16).  p points at (col, c0).
static __device__ __forceinline__ v16bf load_bfrag(const __bf16* p) {
  v8bf lo = *(const v8bf*)p;
  v8bf hi = *(const v8bf*)(p + 16);
  return CAT16(lo, hi);
}

static __device__ __forceinline__ v8bf cvt8(float4 a, float4 b) {
  v8bf r;
  r[0] = (__bf16)a.x; r[1] = (__bf16)a.y; r[2] = (__bf16)a.z; r[3] = (__bf16)a.w;
  r[4] = (__bf16)b.x; r[5] = (__bf16)b.y; r[6] = (__bf16)b.z; r[7] = (__bf16)b.w;
  return r;
}

// ---------------------------------------------------------------------------
// Kernel 1: QKV projection.  out[m,n] = sum_k x[m,k] * W[n,k]  (W row-major NxK)
// blockIdx.z: 0->Q, 1->K (layout B,H,T,HS bf16), 2->V transposed (B,H,HS,T bf16)
// 256 threads = 8 waves in 4x2; block tile 128x128, K-step 32, LDS-staged bf16.
// ---------------------------------------------------------------------------
__global__ __launch_bounds__(256) void qkv_gemm_kernel(
    const float* __restrict__ X, const float* __restrict__ Wq,
    const float* __restrict__ Wk, const float* __restrict__ Wv,
    __bf16* __restrict__ Qo, __bf16* __restrict__ Ko, __bf16* __restrict__ Vto) {
  __shared__ __attribute__((aligned(16))) __bf16 As[128 * 40];  // 128 rows, stride 40 (pad)
  __shared__ __attribute__((aligned(16))) __bf16 Bs[128 * 40];

  const int z = blockIdx.z;
  const float* W = (z == 0) ? Wq : (z == 1 ? Wk : Wv);

  const int tid  = threadIdx.x;
  const int lane = tid & 31;
  const int w    = tid >> 5;
  const int wr   = w >> 1;       // 0..3 -> 32-row slab
  const int wc   = w & 1;        // 0..1 -> 64-col slab
  const int l16  = lane & 15;
  const int hi   = lane >> 4;
  const int c0   = hi * 8;       // A-frag K offset
  const int kb0l = hi * 16;      // unused (B-frag also uses c0)
  const int mBase = blockIdx.y * 128;
  const int nBase = blockIdx.x * 128;

  const int lr = tid >> 1;            // 0..127: LDS row this thread fills
  const int lh = (tid & 1) * 16;      // 0/16 : K half it fills

  const float* aSrc = X + (size_t)(mBase + lr) * Dm + lh;
  const float* bSrc = W + (size_t)(nBase + lr) * Dm + lh;
  __bf16* aDst = &As[lr * 40 + lh];
  __bf16* bDst = &Bs[lr * 40 + lh];

  v8f acc[2][4];
  v8f vz = {};
#pragma unroll
  for (int a = 0; a < 2; ++a)
#pragma unroll
    for (int c = 0; c < 4; ++c) acc[a][c] = vz;

  for (int k0 = 0; k0 < Dm; k0 += 32) {
    __syncthreads();
    {
      const float4* a4 = (const float4*)(aSrc + k0);
      *(v8bf*)aDst       = cvt8(a4[0], a4[1]);
      *(v8bf*)(aDst + 8) = cvt8(a4[2], a4[3]);
      const float4* b4 = (const float4*)(bSrc + k0);
      *(v8bf*)bDst       = cvt8(b4[0], b4[1]);
      *(v8bf*)(bDst + 8) = cvt8(b4[2], b4[3]);
      __builtin_prefetch(aSrc + k0 + 32, 0, 3);   // global_prefetch_b8
      __builtin_prefetch(bSrc + k0 + 32, 0, 3);
    }
    __syncthreads();
    v16bf af[2], bfv[4];
#pragma unroll
    for (int rt = 0; rt < 2; ++rt)
      af[rt] = load_afrag(&As[(wr * 32 + rt * 16 + l16) * 40 + c0]);
#pragma unroll
    for (int ct = 0; ct < 4; ++ct)
      bfv[ct] = load_bfrag(&Bs[(wc * 64 + ct * 16 + l16) * 40 + c0]);
#pragma unroll
    for (int rt = 0; rt < 2; ++rt)
#pragma unroll
      for (int ct = 0; ct < 4; ++ct)
        acc[rt][ct] = wmma_bf16(af[rt], bfv[ct], acc[rt][ct]);
  }

  __bf16* dst = (z == 0) ? Qo : (z == 1 ? Ko : Vto);
#pragma unroll
  for (int rt = 0; rt < 2; ++rt)
#pragma unroll
    for (int ct = 0; ct < 4; ++ct)
#pragma unroll
      for (int i = 0; i < 8; ++i) {
        const int m  = mBase + wr * 32 + rt * 16 + i + 8 * hi;  // C layout: row=i+8*hi
        const int n  = nBase + wc * 64 + ct * 16 + l16;         //           col=l16
        const int bb = m >> 11, tt = m & (Tseq - 1);
        const int hh = n >> 7,  hs = n & (HSd - 1);
        const __bf16 v = (__bf16)acc[rt][ct][i];
        if (z == 2)
          dst[(((size_t)bb * Hn + hh) * HSd + hs) * Tseq + tt] = v;   // V^T
        else
          dst[(((size_t)bb * Hn + hh) * Tseq + tt) * HSd + hs] = v;   // Q, K
      }
}

// ---------------------------------------------------------------------------
// Kernel 2: flash attention, causal.  One wave = 16 queries; block = 128 queries
// of one (b,h).  Q,K in (B,H,T,HS) bf16; V transposed (B,H,HS,T) bf16.
// Per 32-key block: S = 8 wmma, online softmax (shfl_xor within 16-lane halves),
// P transposed C->A layout through wave-private LDS, O += 8 wmma.
// ---------------------------------------------------------------------------
__global__ __launch_bounds__(256) void attn_fa_kernel(
    const __bf16* __restrict__ Q, const __bf16* __restrict__ K,
    const __bf16* __restrict__ Vt, __bf16* __restrict__ Oout) {
  __shared__ __attribute__((aligned(16))) __bf16 pS[8][16][32];  // per-wave P slab

  const int lane = threadIdx.x & 31;
  const int w    = threadIdx.x >> 5;
  const int l16  = lane & 15;
  const int hi   = lane >> 4;
  const int c0   = hi * 8;
  const int kb0l = hi * 16;
  const int bh = blockIdx.y;                 // b*H + h
  const int q0 = blockIdx.x * 128 + w * 16;  // this wave's first query

  const __bf16* Qp = Q  + (size_t)bh * Tseq * HSd;
  const __bf16* Kp = K  + (size_t)bh * Tseq * HSd;
  const __bf16* Vp = Vt + (size_t)bh * HSd * Tseq;

  v16bf qf[4];  // 16x128 Q slab as 4 A-fragments (resident)
  {
    const __bf16* qr = Qp + (size_t)(q0 + l16) * HSd + c0;
#pragma unroll
    for (int j = 0; j < 4; ++j) qf[j] = load_afrag(qr + j * 32);
  }

  float mrow[8], lrow[8];
  v8f o[8];  // 16x128 O accumulator (C layout, 8 hs-tiles)
  v8f vz = {};
#pragma unroll
  for (int i = 0; i < 8; ++i) { mrow[i] = -3.0e38f; lrow[i] = 0.f; o[i] = vz; }

  const float scale = 0.0883883476483184f;  // 1/sqrt(128)
  const int nkb = ((q0 + 15) >> 5) + 1;     // causal: key blocks of 32

  for (int kb = 0; kb < nkb; ++kb) {
    const int kbase = kb << 5;
    // ---- S = Q * K^T (two 16-key column tiles) ----
    v8f s0 = vz, s1 = vz;
    {
      const __bf16* k0p = Kp + (size_t)(kbase + l16) * HSd + c0;
      const __bf16* k1p = k0p + 16 * HSd;
#pragma unroll
      for (int j = 0; j < 4; ++j) {
        s0 = wmma_bf16(qf[j], load_bfrag(k0p + j * 32), s0);
        s1 = wmma_bf16(qf[j], load_bfrag(k1p + j * 32), s1);
      }
    }
    // ---- scale + causal mask + online softmax ----
    const bool need_mask = (kbase + 31) > q0;
    float p0[8], p1[8];
#pragma unroll
    for (int i = 0; i < 8; ++i) {
      const int qrow = q0 + i + 8 * hi;
      float v0 = s0[i] * scale;
      float v1 = s1[i] * scale;
      if (need_mask) {
        if (kbase + l16 > qrow)      v0 = -3.0e38f;
        if (kbase + 16 + l16 > qrow) v1 = -3.0e38f;
      }
      float mx = fmaxf(v0, v1);  // row-max across the 16-lane half
      mx = fmaxf(mx, __shfl_xor(mx, 1, 32));
      mx = fmaxf(mx, __shfl_xor(mx, 2, 32));
      mx = fmaxf(mx, __shfl_xor(mx, 4, 32));
      mx = fmaxf(mx, __shfl_xor(mx, 8, 32));
      const float mnew = fmaxf(mrow[i], mx);
      const float corr = __expf(mrow[i] - mnew);
      const float e0 = __expf(v0 - mnew);
      const float e1 = __expf(v1 - mnew);
      float rs = e0 + e1;
      rs += __shfl_xor(rs, 1, 32);
      rs += __shfl_xor(rs, 2, 32);
      rs += __shfl_xor(rs, 4, 32);
      rs += __shfl_xor(rs, 8, 32);
      lrow[i] = lrow[i] * corr + rs;
      mrow[i] = mnew;
#pragma unroll
      for (int ht = 0; ht < 8; ++ht) o[ht][i] *= corr;
      p0[i] = e0; p1[i] = e1;
    }
    // ---- transpose P (C layout -> A-fragment) through wave-private LDS ----
#pragma unroll
    for (int i = 0; i < 8; ++i) {
      pS[w][i + 8 * hi][l16]      = (__bf16)p0[i];
      pS[w][i + 8 * hi][16 + l16] = (__bf16)p1[i];
    }
    v16bf pf = load_afrag(&pS[w][l16][c0]);
    // ---- O += P * V (V stored transposed: B-frag loads are contiguous) ----
#pragma unroll
    for (int ht = 0; ht < 8; ++ht) {
      const __bf16* vp = Vp + (size_t)(ht * 16 + l16) * Tseq + kbase + c0;
      o[ht] = wmma_bf16(pf, load_bfrag(vp), o[ht]);
    }
  }

  // ---- normalize and store to (B,T,D) bf16 ----
  const int b = bh >> 4;   // H=16
  const int h = bh & 15;
#pragma unroll
  for (int i = 0; i < 8; ++i) {
    const float inv = 1.0f / lrow[i];
    const int tq = q0 + i + 8 * hi;
    __bf16* orow = Oout + ((size_t)b * Tseq + tq) * Dm + h * HSd;
#pragma unroll
    for (int ht = 0; ht < 8; ++ht)
      orow[ht * 16 + l16] = (__bf16)(o[ht][i] * inv);
  }
}

// ---------------------------------------------------------------------------
// Kernel 3: output projection.  out[m,n] = sum_k attn[m,k] * wo[n,k], f32 out.
// ---------------------------------------------------------------------------
__global__ __launch_bounds__(256) void oproj_gemm_kernel(
    const __bf16* __restrict__ Ain, const float* __restrict__ Wo,
    float* __restrict__ Out) {
  __shared__ __attribute__((aligned(16))) __bf16 As[128 * 40];
  __shared__ __attribute__((aligned(16))) __bf16 Bs[128 * 40];

  const int tid  = threadIdx.x;
  const int lane = tid & 31;
  const int w    = tid >> 5;
  const int wr   = w >> 1, wc = w & 1;
  const int l16  = lane & 15, hi = lane >> 4;
  const int c0   = hi * 8;
  const int kb0l = hi * 16;
  const int mBase = blockIdx.y * 128;
  const int nBase = blockIdx.x * 128;
  const int lr = tid >> 1;
  const int lh = (tid & 1) * 16;

  const __bf16* aSrc = Ain + (size_t)(mBase + lr) * Dm + lh;
  const float*  bSrc = Wo  + (size_t)(nBase + lr) * Dm + lh;
  __bf16* aDst = &As[lr * 40 + lh];
  __bf16* bDst = &Bs[lr * 40 + lh];

  v8f acc[2][4];
  v8f vz = {};
#pragma unroll
  for (int a = 0; a < 2; ++a)
#pragma unroll
    for (int c = 0; c < 4; ++c) acc[a][c] = vz;

  for (int k0 = 0; k0 < Dm; k0 += 32) {
    __syncthreads();
    {
      *(v8bf*)aDst       = *(const v8bf*)(aSrc + k0);
      *(v8bf*)(aDst + 8) = *(const v8bf*)(aSrc + k0 + 8);
      const float4* b4 = (const float4*)(bSrc + k0);
      *(v8bf*)bDst       = cvt8(b4[0], b4[1]);
      *(v8bf*)(bDst + 8) = cvt8(b4[2], b4[3]);
      __builtin_prefetch(aSrc + k0 + 32, 0, 3);
      __builtin_prefetch(bSrc + k0 + 32, 0, 3);
    }
    __syncthreads();
    v16bf af[2], bfv[4];
#pragma unroll
    for (int rt = 0; rt < 2; ++rt)
      af[rt] = load_afrag(&As[(wr * 32 + rt * 16 + l16) * 40 + c0]);
#pragma unroll
    for (int ct = 0; ct < 4; ++ct)
      bfv[ct] = load_bfrag(&Bs[(wc * 64 + ct * 16 + l16) * 40 + c0]);
#pragma unroll
    for (int rt = 0; rt < 2; ++rt)
#pragma unroll
      for (int ct = 0; ct < 4; ++ct)
        acc[rt][ct] = wmma_bf16(af[rt], bfv[ct], acc[rt][ct]);
  }

#pragma unroll
  for (int rt = 0; rt < 2; ++rt)
#pragma unroll
    for (int ct = 0; ct < 4; ++ct)
#pragma unroll
      for (int i = 0; i < 8; ++i) {
        const int m = mBase + wr * 32 + rt * 16 + i + 8 * hi;
        const int n = nBase + wc * 64 + ct * 16 + l16;
        Out[(size_t)m * Dm + n] = acc[rt][ct][i];
      }
}

// ---------------------------------------------------------------------------
// Host entry
// ---------------------------------------------------------------------------
extern "C" void kernel_launch(void* const* d_in, const int* in_sizes, int n_in,
                              void* d_out, int out_size, void* d_ws, size_t ws_size,
                              hipStream_t stream) {
  const float* x  = (const float*)d_in[0];
  const float* wq = (const float*)d_in[1];
  const float* wk = (const float*)d_in[2];
  const float* wv = (const float*)d_in[3];
  const float* wo = (const float*)d_in[4];
  float* out = (float*)d_out;

  // workspace: Q | K | V^T | attn  (each B*H*T*HS bf16 = 16 MB, total 64 MB)
  const size_t per = (size_t)Bsz * Hn * Tseq * HSd;
  __bf16* qb  = (__bf16*)d_ws;
  __bf16* kb  = qb + per;
  __bf16* vtb = kb + per;
  __bf16* ab  = vtb + per;

  dim3 blk(256, 1, 1);
  qkv_gemm_kernel<<<dim3(Dm / 128, (Bsz * Tseq) / 128, 3), blk, 0, stream>>>(
      x, wq, wk, wv, qb, kb, vtb);
  attn_fa_kernel<<<dim3(Tseq / 128, Bsz * Hn), blk, 0, stream>>>(qb, kb, vtb, ab);
  oproj_gemm_kernel<<<dim3(Dm / 128, (Bsz * Tseq) / 128), blk, 0, stream>>>(ab, wo, out);
}
